// Kmeans_42271068127238
// MI455X (gfx1250) — matrix operand order repaired and checked by reading, not compiled, not run on hardware
//
#include <hip/hip_runtime.h>
#include <stdint.h>

// ---------------------------------------------------------------------------
// Problem constants (from reference): feature_batch (256,14,14,512) f32
// ---------------------------------------------------------------------------
#define BATCH 256
#define HH    14
#define WW    14
#define CHN   512
#define HW    196                    // 14*14
#define PER_B (HW * CHN)             // 100352 floats per batch
#define TOTAL ((size_t)BATCH * PER_B)

typedef __attribute__((ext_vector_type(16))) _Float16 v16h;
typedef __attribute__((ext_vector_type(8)))  float    v8f;
typedef __attribute__((ext_vector_type(4)))  float    v4f;   // clang vector: OK for nontemporal builtins

// ---------------------------------------------------------------------------
// Kernel A: per-(b,c) spatial argmax -> coords (row, col) as float2
// One block per batch, 128 threads, each owning 4 consecutive channels.
// Loads are b128 and fully coalesced (one 2KB row of 512 channels / step).
// Regular (RT) loads on purpose: input (102.8MB) fits in the 192MB L2,
// so kernel C's re-read hits L2 instead of HBM.
// ---------------------------------------------------------------------------
__global__ __launch_bounds__(128) void argmax_coords(
    const float* __restrict__ in, float* __restrict__ pts /* [B*CHN*2] */) {
  const int b = blockIdx.x;
  const int t = threadIdx.x;                       // 128 threads -> 4 ch each
  const v4f* row = (const v4f*)(in + (size_t)b * PER_B) + t;

  v4f best = row[0];
  int ix = 0, iy = 0, iz = 0, iw = 0;
#pragma unroll 4
  for (int p = 1; p < HW; ++p) {
    v4f v = row[(size_t)p * (CHN / 4)];
    if (v.x > best.x) { best.x = v.x; ix = p; }    // strict > : first-max tie
    if (v.y > best.y) { best.y = v.y; iy = p; }
    if (v.z > best.z) { best.z = v.z; iz = p; }
    if (v.w > best.w) { best.w = v.w; iw = p; }
  }
  const int c = t * 4;
  float* o = pts + ((size_t)b * CHN + c) * 2;
  o[0] = (float)(ix / WW); o[1] = (float)(ix % WW);
  o[2] = (float)(iy / WW); o[3] = (float)(iy % WW);
  o[4] = (float)(iz / WW); o[5] = (float)(iz % WW);
  o[6] = (float)(iw / WW); o[7] = (float)(iw % WW);
}

// ---------------------------------------------------------------------------
// Threefry2x32 (20 rounds) — deterministic centroid seeding.
// ---------------------------------------------------------------------------
__device__ __forceinline__ uint32_t rotl32(uint32_t v, int n) {
  return (v << n) | (v >> (32 - n));
}
__device__ __forceinline__ uint2 threefry2x32(uint32_t k0, uint32_t k1,
                                              uint32_t c0, uint32_t c1) {
  const uint32_t ks2 = k0 ^ k1 ^ 0x1BD11BDAu;
  uint32_t x0 = c0 + k0, x1 = c1 + k1;
  const int RA[4] = {13, 15, 26, 6};
  const int RB[4] = {17, 29, 16, 24};
#define TF_FOUR(R)                                            \
  {                                                           \
    for (int r = 0; r < 4; ++r) {                             \
      x0 += x1; x1 = rotl32(x1, R[r]); x1 ^= x0;              \
    }                                                         \
  }
  TF_FOUR(RA); x0 += k1;  x1 += ks2 + 1u;
  TF_FOUR(RB); x0 += ks2; x1 += k0 + 2u;
  TF_FOUR(RA); x0 += k0;  x1 += k1 + 3u;
  TF_FOUR(RB); x0 += k1;  x1 += ks2 + 4u;
  TF_FOUR(RA); x0 += ks2; x1 += k0 + 5u;
#undef TF_FOUR
  return make_uint2(x0, x1);
}

// ---------------------------------------------------------------------------
// Kernel B: per-batch 2-means, ONE WAVE (32 lanes) per batch.
// The centroid-update reduction sums = onehot^T(2x512) @ [x y 1](512x3) is
// computed with v_wmma_f32_16x16x32_f16 (exact: onehot in {0,1}, coords in
// 0..13, f32 accumulate). 16 WMMAs per iteration, VGPR layouts used here (contiguous K, same in A and B):
//   A 16x32 f16 : lanes 0-15 row M=lane hold K 0..15, lanes 16-31
//                 hold K 16..31. Only rows M=0,1 (clusters) nonzero.
//   B 32x16 f16 : lane l<16 col N=l holds K=0..15, lane l+16 holds K=16..31.
//                 Only cols 0(x),1(y),2(ones) nonzero.
//   D 16x16 f32 : lane N holds column N, vector elem k = row M=k.
// ---------------------------------------------------------------------------
__global__ __launch_bounds__(32) void kmeans_wave(
    const float* __restrict__ pts, uint32_t* __restrict__ bitmask) {
  const int b    = blockIdx.x;
  const int lane = threadIdx.x;

  __shared__ float px[CHN];
  __shared__ float py[CHN];
  __shared__ unsigned char asn[CHN];

#pragma unroll
  for (int j = 0; j < 16; ++j) {
    const int p = lane + 32 * j;
    px[p] = pts[((size_t)b * CHN + p) * 2 + 0];
    py[p] = pts[((size_t)b * CHN + p) * 2 + 1];
  }
  __syncthreads();

  // Seed two distinct centroid indices (Threefry surrogate for jax perm).
  const uint2 kb = threefry2x32(0u, 42u, (uint32_t)(2 * b), (uint32_t)(2 * b + 1));
  const uint2 rr = threefry2x32(kb.x, kb.y, 0u, 1u);
  const int i0 = (int)(rr.x & 511u);
  int i1 = (int)(rr.y % 511u); if (i1 >= i0) ++i1;
  float cx0 = px[i0], cy0 = py[i0];
  float cx1 = px[i1], cy1 = py[i1];

  bool la[16];

  for (int it = 0; it <= 10; ++it) {
    // ---- assignment: lane owns points p = lane + 32*j (strided) ----
#pragma unroll
    for (int j = 0; j < 16; ++j) {
      const int p = lane + 32 * j;
      const float dx0 = px[p] - cx0, dy0 = py[p] - cy0;
      const float dx1 = px[p] - cx1, dy1 = py[p] - cy1;
      const bool a1 = (dx1 * dx1 + dy1 * dy1) < (dx0 * dx0 + dy0 * dy0);
      la[j]  = a1;                   // argmin ties -> cluster 0 (first index)
      asn[p] = a1 ? 1 : 0;
    }
    __syncthreads();
    if (it == 10) break;             // 11th assignment: no more updates

    // ---- sums/counts via WMMA over 16 chunks of 32 points ----
    v8f acc = {0.f, 0.f, 0.f, 0.f, 0.f, 0.f, 0.f, 0.f};
    const bool fillA = (lane == 0) || (lane == 1) || (lane == 16) || (lane == 17);
    const int  kb0   = (lane & 16) ? 16 : 0;  // A half-row K base
    const int  cl    = lane & 1;              // cluster row for A
    const int  ncol  = lane & 15;             // B column index
    const int  koffB = (lane & 16) ? 16 : 0;  // B K base

    for (int ch = 0; ch < 16; ++ch) {
      const int base = ch * 32;
      v16h A  = {};
      v16h Bm = {};
      if (fillA) {
#pragma unroll
        for (int e = 0; e < 8; ++e) {
          A[e]     = (_Float16)((asn[base + kb0 + e]      == cl) ? 1.0f : 0.0f);
          A[e + 8] = (_Float16)((asn[base + kb0 + 8 + e] == cl) ? 1.0f : 0.0f);
        }
      }
      if (ncol == 0) {
#pragma unroll
        for (int e = 0; e < 16; ++e) Bm[e] = (_Float16)px[base + koffB + e];
      } else if (ncol == 1) {
#pragma unroll
        for (int e = 0; e < 16; ++e) Bm[e] = (_Float16)py[base + koffB + e];
      } else if (ncol == 2) {
#pragma unroll
        for (int e = 0; e < 16; ++e) Bm[e] = (_Float16)1.0f;
      }
      acc = __builtin_amdgcn_wmma_f32_16x16x32_f16(
          false, A, false, Bm, (short)0, acc, false, false);
    }

    // ---- extract D[k][n]: lane n, elem k ----
    const float sx0 = __shfl(acc[0], 0, 32);
    const float sy0 = __shfl(acc[0], 1, 32);
    const float n0  = __shfl(acc[0], 2, 32);
    const float sx1 = __shfl(acc[1], 0, 32);
    const float sy1 = __shfl(acc[1], 1, 32);
    const float n1  = __shfl(acc[1], 2, 32);
    const float r0 = 1.0f / fmaxf(n0, 1.0f);
    const float r1 = 1.0f / fmaxf(n1, 1.0f);
    cx0 = sx0 * r0; cy0 = sy0 * r0;
    cx1 = sx1 * r1; cy1 = sy1 * r1;
    __syncthreads();                 // protect asn WAR for next iteration
  }

  // ---- pack per-batch 512-bit assignment mask with wave32 ballot ----
#pragma unroll
  for (int j = 0; j < 16; ++j) {
#if defined(__has_builtin) && __has_builtin(__builtin_amdgcn_ballot_w32)
    const uint32_t w = __builtin_amdgcn_ballot_w32(la[j]);
#else
    const uint32_t w = (uint32_t)__ballot(la[j]);
#endif
    if (lane == 0) bitmask[b * 16 + j] = w;
  }
}

// ---------------------------------------------------------------------------
// Kernel C: masked dual-copy (the bandwidth kernel, ~308MB of HBM traffic).
// 2 blocks per batch, 256 threads. Input staged through LDS with CDNA5
// async-to-LDS b128 loads, 2-deep double buffer, s_wait_asynccnt pipelining.
// Input re-read hits L2 (warmed by kernel A); outputs stored non-temporal so
// 205MB of write-once data does not evict the cached input.
// Tile stride 1024 == 2*CHN, so each thread's channel (and mask bits) are
// loop-invariant.
// ---------------------------------------------------------------------------
__global__ __launch_bounds__(256) void apply_mask(
    const float* __restrict__ in, const uint32_t* __restrict__ bitmask,
    float* __restrict__ out0, float* __restrict__ out1) {
  const int blk  = blockIdx.x;         // 512 blocks: 2 per batch
  const int b    = blk >> 1;
  const int half = blk & 1;
  const int tid  = threadIdx.x;

  __shared__ float    stage[2][1024];  // 2 x 4KB double buffer
  __shared__ uint32_t mask[16];

  if (tid < 16) mask[tid] = bitmask[b * 16 + tid];
  __syncthreads();

  const size_t base = (size_t)b * PER_B + (size_t)half * (PER_B / 2);
  const int NT = (PER_B / 2) / 1024;   // 49 tiles of 1024 floats

  // Loop-invariant channel + mask bits for this thread's 4 floats.
  const int e = (tid * 4) & (CHN - 1);
  const uint32_t bits = (mask[e >> 5] >> (e & 31)) & 0xFu;

  // LDS byte addresses of this thread's 16B slot in each buffer
  // (flat-address low 32 bits == LDS offset on gfx1250).
  const uint32_t lds0 = (uint32_t)(uintptr_t)&stage[0][tid * 4];
  const uint32_t lds1 = (uint32_t)(uintptr_t)&stage[1][tid * 4];

  // Prologue: async-load tile 0 into buffer 0.
  {
    const float* g = in + base + (size_t)tid * 4;
    asm volatile("global_load_async_to_lds_b128 %0, %1, off"
                 :: "v"(lds0), "v"(g) : "memory");
  }

  for (int t = 0; t < NT; ++t) {
    const int buf = t & 1;
    if (t + 1 < NT) {
      const float* g = in + base + (size_t)(t + 1) * 1024 + (size_t)tid * 4;
      asm volatile("global_load_async_to_lds_b128 %0, %1, off"
                   :: "v"(buf ? lds0 : lds1), "v"(g) : "memory");
      asm volatile("s_wait_asynccnt 1" ::: "memory");   // tile t landed
    } else {
      asm volatile("s_wait_asynccnt 0" ::: "memory");   // drain
    }

    const v4f v = *(const v4f*)&stage[buf][tid * 4];
    v4f o0, o1;
    o0.x = (bits & 1u) ? 0.0f : v.x;  o1.x = (bits & 1u) ? v.x : 0.0f;
    o0.y = (bits & 2u) ? 0.0f : v.y;  o1.y = (bits & 2u) ? v.y : 0.0f;
    o0.z = (bits & 4u) ? 0.0f : v.z;  o1.z = (bits & 4u) ? v.z : 0.0f;
    o0.w = (bits & 8u) ? 0.0f : v.w;  o1.w = (bits & 8u) ? v.w : 0.0f;

    const size_t off = base + (size_t)t * 1024 + (size_t)tid * 4;
    __builtin_nontemporal_store(o0, (v4f*)(out0 + off));
    __builtin_nontemporal_store(o1, (v4f*)(out1 + off));
  }
}

// ---------------------------------------------------------------------------
// Launch: A (argmax -> coords in ws) -> B (k-means -> bitmask in ws) -> C.
// Workspace layout: [ coords: 256*512 float2 = 1MB | bitmask: 256*16 u32 ].
// ---------------------------------------------------------------------------
extern "C" void kernel_launch(void* const* d_in, const int* in_sizes, int n_in,
                              void* d_out, int out_size, void* d_ws,
                              size_t ws_size, hipStream_t stream) {
  (void)in_sizes; (void)n_in; (void)out_size; (void)ws_size;
  const float* in = (const float*)d_in[0];
  float*    pts = (float*)d_ws;
  uint32_t* bm  = (uint32_t*)((char*)d_ws +
                              (size_t)BATCH * CHN * 2 * sizeof(float));
  float* out0 = (float*)d_out;
  float* out1 = out0 + TOTAL;

  argmax_coords<<<BATCH, 128, 0, stream>>>(in, pts);
  kmeans_wave  <<<BATCH, 32, 0, stream>>>(pts, bm);
  apply_mask   <<<BATCH * 2, 256, 0, stream>>>(in, bm, out0, out1);
}
